// QuantizedMultiheadAttention_73684458930262
// MI455X (gfx1250) — matrix operand order repaired and checked by reading, not compiled, not run on hardware
//
#include <hip/hip_runtime.h>
#include <hip/hip_bf16.h>
#include <stdint.h>

// ---------------------------------------------------------------------------
// QuantizedMultiheadAttention for MI455X (gfx1250, wave32, WMMA).
// quantize() == saturating f32->f16 round-to-nearest  =>  all matmuls are
// v_wmma_f32_16x16x32_f16 with quantize fused into LDS staging / C store.
// Scores materialize straight into the attn_weights region of d_out (it is a
// required output, so flash-attention buys nothing), softmax in place, then
// re-read (quantized to f16 on stage) for the P@V GEMM.
// Pure f16 copy staging uses GLOBAL_LOAD_ASYNC_TO_LDS_B128 (ASYNCcnt).
// ---------------------------------------------------------------------------

typedef __attribute__((ext_vector_type(16))) _Float16 v16h;
typedef __attribute__((ext_vector_type(8)))  _Float16 v8h;
typedef __attribute__((ext_vector_type(8)))  float    v8f;

__device__ __forceinline__ _Float16 quant_h(float x) {
    // fp16 fake-quant: saturate to +-65504, RNE on convert
    x = fminf(fmaxf(x, -65504.0f), 65504.0f);
    return (_Float16)x;
}

template <typename OT, bool QOUT>
__device__ __forceinline__ void store_c(OT* p, float v) {
    if constexpr (sizeof(OT) == 2) {
        *p = (OT)quant_h(v);
    } else {
        if constexpr (QOUT) *p = (float)quant_h(v);
        else                *p = v;
    }
}

// 16-byte async global->LDS copy (GVS mode: SGPR64 base + VGPR32 byte offset).
// LDS dest offset comes from the low 32 bits of the generic shared pointer
// (ISA: LDS aperture addr[31:0] is the workgroup-relative LDS byte address).
__device__ __forceinline__ void async_copy16(uint32_t lds_off,
                                             const void* gbase,
                                             uint32_t goff_bytes) {
    asm volatile("global_load_async_to_lds_b128 %0, %1, %2"
                 :: "v"(lds_off), "v"(goff_bytes), "s"(gbase)
                 : "memory");
}
__device__ __forceinline__ void wait_async0() {
    asm volatile("s_wait_asynccnt 0" ::: "memory");
}

// ---------------------------------------------------------------------------
// Generic WMMA GEMM:  C[z] = op( A[z] (MxK row-major) * B[z] (KxN) ) * scale
//   AT     : float (quantize->f16 on stage) or _Float16 (async copy on stage)
//   BMODE  : 0 = B f32 KxN row-major, quantize + transpose-stage  (weights)
//            1 = B f16 already N-major ("Bt": Bt[n*ldb + k]) async copy
//            2 = B f16 KxN row-major, transpose-stage              (V for P@V)
//   per-z base offsets: z -> (b = z/zH, h = z%zH), base = b*sb + h*sh
// M,N assumed multiples of BM,BN; K multiple of BK (true for all launches).
// ---------------------------------------------------------------------------
template <int BM, int BN, int BK, int WM, int WN,
          typename AT, int BMODE, typename OT, bool QOUT>
__global__ __launch_bounds__(256) void gemm_wmma_kernel(
    const AT* __restrict__ A, int lda, long long asb, long long ash,
    const void* __restrict__ Bv, int ldb, long long bsb, long long bsh,
    OT* __restrict__ C, int ldc, long long csb, long long csh,
    int K, int zH, float scale)
{
    constexpr int LDA  = BK + 8;          // halfs; keeps 16B alignment (+ skews banks)
    constexpr int LDB  = BK + 8;
    constexpr int WTM  = BM / WM;
    constexpr int WTN  = BN / WN;
    constexpr int MSUB = WTM / 16;
    constexpr int NSUB = WTN / 16;
    static_assert(WM * WN == 8, "8 waves per block");

    __shared__ __align__(16) _Float16 sA[BM * LDA];
    __shared__ __align__(16) _Float16 sB[BN * LDB];

    const int z  = blockIdx.z;
    const int bz = z / zH, hz = z % zH;
    const AT* Ab = A + (long long)bz * asb + (long long)hz * ash;
    OT*       Cb = C + (long long)bz * csb + (long long)hz * csh;
    const float*    B32b = (const float*)Bv    + (long long)bz * bsb + (long long)hz * bsh;
    const _Float16* B16b = (const _Float16*)Bv + (long long)bz * bsb + (long long)hz * bsh;

    const int tileM = blockIdx.y * BM;
    const int tileN = blockIdx.x * BN;

    const int tid   = threadIdx.x;
    const int wave  = tid >> 5;
    const int lane  = tid & 31;
    const int waveM = wave % WM;
    const int waveN = wave / WM;
    const int m16   = lane & 15;
    const int kg    = lane >> 4;   // half-wave group

    v8f acc[MSUB][NSUB];
#pragma unroll
    for (int i = 0; i < MSUB; ++i)
#pragma unroll
        for (int j = 0; j < NSUB; ++j)
#pragma unroll
            for (int e = 0; e < 8; ++e) acc[i][j][e] = 0.0f;

    for (int k0 = 0; k0 < K; k0 += BK) {
        // ---- stage A tile (BM x BK) into LDS, f16 ----
        constexpr int ACH = (BM * BK) / 8;   // 8-half (16B) chunks
        for (int c = tid; c < ACH; c += 256) {
            const int r  = (c * 8) / BK;
            const int cc = (c * 8) % BK;
            if constexpr (sizeof(AT) == 4) {
                const float4* src =
                    (const float4*)(Ab + (size_t)(tileM + r) * lda + k0 + cc);
                float4 f0 = src[0], f1 = src[1];
                v8h hv;
                hv[0] = quant_h(f0.x); hv[1] = quant_h(f0.y);
                hv[2] = quant_h(f0.z); hv[3] = quant_h(f0.w);
                hv[4] = quant_h(f1.x); hv[5] = quant_h(f1.y);
                hv[6] = quant_h(f1.z); hv[7] = quant_h(f1.w);
                *(v8h*)(&sA[r * LDA + cc]) = hv;
            } else {
                async_copy16((uint32_t)(uintptr_t)&sA[r * LDA + cc], Ab,
                             (uint32_t)(((size_t)(tileM + r) * lda + k0 + cc) *
                                        sizeof(_Float16)));
            }
        }
        // ---- stage B tile into LDS as Bt[n][k] (BN x BK), f16 ----
        if constexpr (BMODE == 1) {
            constexpr int BCH = (BN * BK) / 8;
            for (int c = tid; c < BCH; c += 256) {
                const int n  = (c * 8) / BK;
                const int kk = (c * 8) % BK;
                async_copy16((uint32_t)(uintptr_t)&sB[n * LDB + kk], B16b,
                             (uint32_t)(((size_t)(tileN + n) * ldb + k0 + kk) *
                                        sizeof(_Float16)));
            }
        } else {
            for (int idx = tid; idx < BK * BN; idx += 256) {
                const int kk = idx / BN;
                const int n  = idx % BN;
                _Float16 hv;
                if constexpr (BMODE == 0)
                    hv = quant_h(B32b[(size_t)(k0 + kk) * ldb + tileN + n]);
                else
                    hv = B16b[(size_t)(k0 + kk) * ldb + tileN + n];
                sB[n * LDB + kk] = hv;
            }
        }
        if constexpr (sizeof(AT) == 2 || BMODE == 1) wait_async0();
        __syncthreads();

        // ---- WMMA over the staged tile ----
#pragma unroll
        for (int kk = 0; kk < BK; kk += 32) {
            v16h af[MSUB], bf[NSUB];
#pragma unroll
            for (int i = 0; i < MSUB; ++i) {
                // A 16-bit 16x32 layout: lanes0-15 K={0..7,16..23}, lanes16-31 +8
                const _Float16* p =
                    &sA[(waveM * WTM + i * 16 + m16) * LDA + kk + kg * 8];
                v8h lo = *(const v8h*)p;
                v8h hi = *(const v8h*)(p + 16);
                af[i] = __builtin_shufflevector(lo, hi, 0,1,2,3,4,5,6,7,
                                                        8,9,10,11,12,13,14,15);
            }
#pragma unroll
            for (int j = 0; j < NSUB; ++j) {
                // B 16-bit 32x16 layout: lanes0-15 K={0..7,16..23}, lanes16-31 +8
                const _Float16* p =
                    &sB[(waveN * WTN + j * 16 + m16) * LDB + kk + kg * 8];
                v8h lo = *(const v8h*)p;
                v8h hi = *(const v8h*)(p + 16);
                bf[j] = __builtin_shufflevector(lo, hi, 0,1,2,3,4,5,6,7,
                                                        8,9,10,11,12,13,14,15);
            }
#pragma unroll
            for (int i = 0; i < MSUB; ++i)
#pragma unroll
                for (int j = 0; j < NSUB; ++j)
                    acc[i][j] = __builtin_amdgcn_wmma_f32_16x16x32_f16(
                        false, af[i], false, bf[j],
                        (short)0, acc[i][j], false, false);
        }
        __syncthreads();
    }

    // ---- store C: c[e] -> row = e + 8*(lane>>4), col = lane&15 ----
    const int mo = kg * 8;
#pragma unroll
    for (int i = 0; i < MSUB; ++i)
#pragma unroll
        for (int j = 0; j < NSUB; ++j) {
            const int row0 = tileM + waveM * WTM + i * 16 + mo;
            const int col  = tileN + waveN * WTN + j * 16 + m16;
#pragma unroll
            for (int e = 0; e < 8; ++e)
                store_c<OT, QOUT>(&Cb[(size_t)(row0 + e) * ldc + col],
                                  acc[i][j][e] * scale);
        }
}

// ---------------------------------------------------------------------------
// In-place row softmax over 1024-wide rows (one block per row, 8 waves).
// ---------------------------------------------------------------------------
__global__ __launch_bounds__(256) void softmax_rows_kernel(float* __restrict__ p)
{
    __shared__ float red[8];
    float* row = p + (size_t)blockIdx.x * 1024;
    const int tid = threadIdx.x;

    float4 v = ((const float4*)row)[tid];
    float m = fmaxf(fmaxf(v.x, v.y), fmaxf(v.z, v.w));
#pragma unroll
    for (int o = 16; o > 0; o >>= 1) m = fmaxf(m, __shfl_xor(m, o, 32));
    if ((tid & 31) == 0) red[tid >> 5] = m;
    __syncthreads();
    float bm = red[0];
#pragma unroll
    for (int i = 1; i < 8; ++i) bm = fmaxf(bm, red[i]);
    __syncthreads();

    v.x = __expf(v.x - bm); v.y = __expf(v.y - bm);
    v.z = __expf(v.z - bm); v.w = __expf(v.w - bm);
    float s = (v.x + v.y) + (v.z + v.w);
#pragma unroll
    for (int o = 16; o > 0; o >>= 1) s += __shfl_xor(s, o, 32);
    if ((tid & 31) == 0) red[tid >> 5] = s;
    __syncthreads();
    float tot = 0.0f;
#pragma unroll
    for (int i = 0; i < 8; ++i) tot += red[i];
    const float inv = 1.0f / tot;

    v.x *= inv; v.y *= inv; v.z *= inv; v.w *= inv;
    ((float4*)row)[tid] = v;
}

// ---------------------------------------------------------------------------
extern "C" void kernel_launch(void* const* d_in, const int* in_sizes, int n_in,
                              void* d_out, int out_size, void* d_ws, size_t ws_size,
                              hipStream_t stream)
{
    (void)in_sizes; (void)n_in; (void)out_size; (void)ws_size;

    constexpr int B = 4, S = 1024, D = 1024, H = 16, HD = 64;
    constexpr long long SD = (long long)S * D;          // 1M elems
    constexpr long long BH_P = (long long)S * S;        // 1M: per-head score plane
    constexpr long long B_P  = (long long)H * BH_P;     // 16M: per-batch score block

    const float* query = (const float*)d_in[0];
    const float* keyi  = (const float*)d_in[1];
    const float* value = (const float*)d_in[2];
    const float* q_w   = (const float*)d_in[3];
    const float* k_w   = (const float*)d_in[4];
    const float* v_w   = (const float*)d_in[5];
    const float* o_w   = (const float*)d_in[6];

    float* out   = (float*)d_out;                       // [B,S,D]
    float* attnw = out + (size_t)B * SD;                // [B,H,S,S] (also scratch)

    // workspace: 4 x 4M f16 activations = 32 MB
    _Float16* q_act = (_Float16*)d_ws;
    _Float16* k_act = q_act + (size_t)B * SD;
    _Float16* v_act = k_act + (size_t)B * SD;
    _Float16* o_act = v_act + (size_t)B * SD;

    const dim3 blk(256);

    // 1) projections: [4096,1024] x [1024,1024] -> f16 (quantized)
    const dim3 gproj(D / 128, (B * S) / 128, 1);
    gemm_wmma_kernel<128,128,64,2,4, float,0,_Float16,true>
        <<<gproj, blk, 0, stream>>>(query, D, 0, 0, q_w, D, 0, 0,
                                    q_act, D, 0, 0, D, 1, 1.0f);
    gemm_wmma_kernel<128,128,64,2,4, float,0,_Float16,true>
        <<<gproj, blk, 0, stream>>>(keyi,  D, 0, 0, k_w, D, 0, 0,
                                    k_act, D, 0, 0, D, 1, 1.0f);
    gemm_wmma_kernel<128,128,64,2,4, float,0,_Float16,true>
        <<<gproj, blk, 0, stream>>>(value, D, 0, 0, v_w, D, 0, 0,
                                    v_act, D, 0, 0, D, 1, 1.0f);

    // 2) scores = Q K^T / 8 -> attn_weights region (raw f32)
    //    per (b,h): M=N=1024, K=64; K-rows row-major == Bt layout for free
    gemm_wmma_kernel<128,128,64,2,4, _Float16,1,float,false>
        <<<dim3(S / 128, S / 128, B * H), blk, 0, stream>>>(
            q_act, D, SD, HD,          // A: Q[b, q, h*64 + d]
            k_act, D, SD, HD,          // Bt: K[b, kj, h*64 + d]
            attnw, S, B_P, BH_P,       // C: P[b, h, q, kj]
            HD, H, 0.125f);

    // 3) softmax in place over last axis (64k rows of 1024)
    softmax_rows_kernel<<<dim3(B * H * S), blk, 0, stream>>>(attnw);

    // 4) O = P V   per (b,h): M=1024, N=64, K=1024 (P f32 -> f16 on stage,
    //    V transpose-staged), result into interleaved [B,S,D] f16
    gemm_wmma_kernel<128,64,64,4,2, float,2,_Float16,true>
        <<<dim3(1, S / 128, B * H), blk, 0, stream>>>(
            attnw, S, B_P, BH_P,       // A: P[b, h, q, kj]
            v_act, D, SD, HD,          // B: V[b, kj, h*64 + d] (KxN)
            o_act, D, SD, HD,          // C: O[b, q, h*64 + d]
            S, H, 1.0f);

    // 5) output = quantize(O @ W_out)  -> f32 out
    gemm_wmma_kernel<128,128,64,2,4, _Float16,0,float,true>
        <<<gproj, blk, 0, stream>>>(o_act, D, 0, 0, o_w, D, 0, 0,
                                    out, D, 0, 0, D, 1, 1.0f);
}
